// MC_Att_L_69088843923954
// MI455X (gfx1250) — matrix operand order repaired and checked by reading, not compiled, not run on hardware
//
#include <hip/hip_runtime.h>
#include <hip/hip_bf16.h>

// ---------------- problem constants ----------------
#define N_NODES 50000
#define N_EDGES 800000
#define IN_NF   128
#define HID     128
#define N_CH    4
#define DD      3
#define KPAD    144      // h row padded to 144 f16 (128 data + 16 zeros)
#define KVK     160      // kv K padded 144 -> 160 (5 steps of 32)
#define COORD_CLAMP 10.0f

typedef __attribute__((ext_vector_type(16))) _Float16 v16h;
typedef __attribute__((ext_vector_type(8)))  _Float16 v8h;
typedef __attribute__((ext_vector_type(8)))  float    v8f;

union V16 { v16h v; v8h h[2]; };

// ---- WMMA wrapper: D = A(16x32 f16) * B(32x16 f16) + C(16x16 f32) ----
__device__ __forceinline__ v8f wmma_f16(v16h a, v16h b, v8f c) {
  return __builtin_amdgcn_wmma_f32_16x16x32_f16(
      /*neg_a=*/false, a, /*neg_b=*/false, b,
      /*c_mod=*/(short)0, c, /*reuse_a=*/false, /*reuse_b=*/false);
}

// A-fragment: src points at (row base + 32*s). ISA 16-bit A layout:
//   lane<16: t0..7 -> K 0..7, t8..15 -> K 16..23
//   lane>=16: t0..7 -> K 8..15, t8..15 -> K 24..31
__device__ __forceinline__ v16h loadA(const _Float16* src, int hi) {
  V16 u;
  u.h[0] = *(const v8h*)(src + (hi ? 8 : 0));
  u.h[1] = *(const v8h*)(src + 16 + (hi ? 8 : 0));
  return u.v;
}

// B-fragment from row-major W^T[n][k]: lane holds N = lane%16,
// K = 32*s + 8*(lane/16) + t (t<8), 32*s + 16 + 8*(lane/16) + (t-8) (t>=8)
__device__ __forceinline__ v16h loadB(const _Float16* rowbase, int s, int hi) {
  const _Float16* p = rowbase + 32 * s + (hi ? 8 : 0);
  V16 u;
  u.h[0] = *(const v8h*)p;
  u.h[1] = *(const v8h*)(p + 16);
  return u.v;
}

__device__ __forceinline__ void atomicMaxF(float* addr, float val) {
  int old = __float_as_int(*addr);
  while (__int_as_float(old) < val) {
    int assumed = old;
    old = atomicCAS((int*)addr, assumed, __float_as_int(val));
    if (old == assumed) break;
  }
}

// ---------------- prep kernels ----------------
__global__ void k_init_nodes(float* nmax, float* nsum) {
  int i = blockIdx.x * 256 + threadIdx.x;
  if (i < N_NODES) { nmax[i] = -3.0e38f; nsum[i] = 0.f; }
}
__global__ void k_zero(float* p, int n) {
  int i = blockIdx.x * 256 + threadIdx.x;
  if (i < n) p[i] = 0.f;
}
__global__ void k_hpad(const float* __restrict__ h, _Float16* __restrict__ hbf) {
  int i = blockIdx.x * 256 + threadIdx.x;
  if (i < N_NODES * KPAD) {
    int node = i / KPAD, j = i % KPAD;
    hbf[i] = (j < IN_NF) ? (_Float16)h[node * IN_NF + j] : (_Float16)0.f;
  }
}
__global__ void k_wq(const float* __restrict__ Wq, _Float16* __restrict__ WqT) {
  int i = blockIdx.x * 256 + threadIdx.x;
  if (i < 128 * 128) { int n = i / 128, k = i % 128; WqT[i] = (_Float16)Wq[k * 128 + n]; }
}
__global__ void k_wkv(const float* __restrict__ Wkv, _Float16* __restrict__ WkT,
                      _Float16* __restrict__ WvT) {
  int i = blockIdx.x * 256 + threadIdx.x;
  if (i < 128 * KVK) {
    int n = i / KVK, k = i % KVK;
    WkT[i] = (k < 144) ? (_Float16)Wkv[k * 256 + 2 * n]     : (_Float16)0.f;
    WvT[i] = (k < 144) ? (_Float16)Wkv[k * 256 + 2 * n + 1] : (_Float16)0.f;
  }
}
__global__ void k_wc1(const float* __restrict__ Wc1, _Float16* __restrict__ Wc1T) {
  int i = blockIdx.x * 256 + threadIdx.x;
  if (i < 256 * 128) { int n = i / 128, k = i % 128; Wc1T[i] = (_Float16)Wc1[k * 256 + n]; }
}
__global__ void k_wc2(const float* __restrict__ Wc2, _Float16* __restrict__ Wc2T) {
  int i = blockIdx.x * 256 + threadIdx.x;
  if (i < 16 * 256) {
    int n = i / 256, k = i % 256;
    Wc2T[i] = (n < 4) ? (_Float16)Wc2[k * 4 + n] : (_Float16)0.f;
  }
}

// ---------------- pass 1: alpha = q.k per edge, seg-max ----------------
__global__ __launch_bounds__(256) void k_alpha(
    const int* __restrict__ ei, const float* __restrict__ coord,
    const _Float16* __restrict__ hbf,
    const _Float16* __restrict__ WqT, const _Float16* __restrict__ WkT,
    const float* __restrict__ bq, const float* __restrict__ bkv,
    float* __restrict__ alphabuf, float* __restrict__ nodemax) {
  int lane = threadIdx.x & 31, wave = threadIdx.x >> 5;
  int m = lane & 15, hi = lane >> 4;
  int tile = blockIdx.x * 8 + wave;
  int ebase = tile * 16, e = ebase + m;
  int ridx = ei[e], cidx = ei[N_EDGES + e];

  float df[12];
#pragma unroll
  for (int i = 0; i < 12; i++) df[i] = coord[ridx * 12 + i] - coord[cidx * 12 + i];
  float rad[8];
#pragma unroll
  for (int a2 = 0; a2 < 2; a2++) {
    int ca = 2 * hi + a2;
#pragma unroll
    for (int cb = 0; cb < 4; cb++)
      rad[a2 * 4 + cb] = df[ca * 3] * df[cb * 3] + df[ca * 3 + 1] * df[cb * 3 + 1] +
                         df[ca * 3 + 2] * df[cb * 3 + 2];
  }
  const _Float16* hr = hbf + (size_t)ridx * KPAD;
  const _Float16* hc = hbf + (size_t)cidx * KPAD;

  v16h aq[4];
#pragma unroll
  for (int s = 0; s < 4; s++) aq[s] = loadA(hr + 32 * s, hi);

  v16h ak[5];
  {
    V16 u;
#pragma unroll
    for (int t = 0; t < 8; t++) u.h[0][t] = (_Float16)rad[t];   // radial -> K 0..15
    u.h[1] = *(const v8h*)(hc + (hi ? 8 : 0));                  // h[col][0..15]
    ak[0] = u.v;
  }
#pragma unroll
  for (int s = 1; s < 5; s++) {
    V16 u;
    u.h[0] = *(const v8h*)(hc + 32 * s - 16 + (hi ? 8 : 0));
    u.h[1] = *(const v8h*)(hc + 32 * s + (hi ? 8 : 0));
    ak[s] = u.v;
  }

  float dot[8];
#pragma unroll
  for (int r = 0; r < 8; r++) dot[r] = 0.f;
#pragma unroll
  for (int j = 0; j < 8; j++) {
    v8f qa = {}; v8f ka = {};
    const _Float16* wqrow = WqT + (16 * j + m) * 128;
#pragma unroll
    for (int s = 0; s < 4; s++) qa = wmma_f16(aq[s], loadB(wqrow, s, hi), qa);
    const _Float16* wkrow = WkT + (16 * j + m) * KVK;
#pragma unroll
    for (int s = 0; s < 5; s++) ka = wmma_f16(ak[s], loadB(wkrow, s, hi), ka);
    int n = 16 * j + m;
    float bqn = bq[n], bkn = bkv[2 * n];
#pragma unroll
    for (int r = 0; r < 8; r++) dot[r] += (qa[r] + bqn) * (ka[r] + bkn);
  }
#pragma unroll
  for (int r = 0; r < 8; r++) {
    float v = dot[r];
    v += __shfl_xor(v, 1); v += __shfl_xor(v, 2);
    v += __shfl_xor(v, 4); v += __shfl_xor(v, 8);
    dot[r] = v;
  }
  int ridr[8];
#pragma unroll
  for (int r = 0; r < 8; r++) ridr[r] = __shfl(ridx, r + 8 * hi);
  if (m == 0) {
#pragma unroll
    for (int r = 0; r < 8; r++) {
      int e2 = ebase + r + 8 * hi;
      alphabuf[e2] = dot[r];
      atomicMaxF(&nodemax[ridr[r]], dot[r]);
    }
  }
}

// ---------------- pass 2: exp(alpha - max), seg-sum ----------------
__global__ void k_softmax_ex(const int* __restrict__ ei, float* __restrict__ alphabuf,
                             const float* __restrict__ nodemax, float* __restrict__ nodesum) {
  int e = blockIdx.x * 256 + threadIdx.x;
  if (e < N_EDGES) {
    int r = ei[e];
    float ex = __expf(alphabuf[e] - nodemax[r]);
    alphabuf[e] = ex;
    atomicAdd(&nodesum[r], ex);
  }
}

// ---------------- pass 3: V, MLP, aggregation ----------------
struct WSv { _Float16 buf[16 * 256]; int rows[16]; float att[16]; float cv[64]; };

__global__ __launch_bounds__(128) void k_aggregate(
    const int* __restrict__ ei, const float* __restrict__ coord,
    const _Float16* __restrict__ hbf,
    const _Float16* __restrict__ WvT, const _Float16* __restrict__ Wc1T,
    const _Float16* __restrict__ Wc2T, const float* __restrict__ bkv,
    const float* __restrict__ exbuf, const float* __restrict__ nodesum,
    float* __restrict__ agg_h, float* __restrict__ agg_c, float* __restrict__ out_att) {
  __shared__ WSv sh[4];
  int lane = threadIdx.x & 31, wave = threadIdx.x >> 5;
  int m = lane & 15, hi = lane >> 4;
  WSv& W = sh[wave];
  int tile = blockIdx.x * 4 + wave;
  int ebase = tile * 16, e = ebase + m;
  int ridx = ei[e], cidx = ei[N_EDGES + e];

  float df[12];
#pragma unroll
  for (int i = 0; i < 12; i++) df[i] = coord[ridx * 12 + i] - coord[cidx * 12 + i];
  float rad[8];
#pragma unroll
  for (int a2 = 0; a2 < 2; a2++) {
    int ca = 2 * hi + a2;
#pragma unroll
    for (int cb = 0; cb < 4; cb++)
      rad[a2 * 4 + cb] = df[ca * 3] * df[cb * 3] + df[ca * 3 + 1] * df[cb * 3 + 1] +
                         df[ca * 3 + 2] * df[cb * 3 + 2];
  }
  if (hi == 0) {
    float att = exbuf[e] / nodesum[ridx];
    W.att[m] = att; W.rows[m] = ridx;
    out_att[e] = att;
  }
  __builtin_amdgcn_wave_barrier();   // same-wave LDS ops are in-order; block reordering
  float attv[8]; int rowsv[8];
#pragma unroll
  for (int r = 0; r < 8; r++) { attv[r] = W.att[r + 8 * hi]; rowsv[r] = W.rows[r + 8 * hi]; }

  const _Float16* hc = hbf + (size_t)cidx * KPAD;
  v16h ak[5];
  {
    V16 u;
#pragma unroll
    for (int t = 0; t < 8; t++) u.h[0][t] = (_Float16)rad[t];
    u.h[1] = *(const v8h*)(hc + (hi ? 8 : 0));
    ak[0] = u.v;
  }
#pragma unroll
  for (int s = 1; s < 5; s++) {
    V16 u;
    u.h[0] = *(const v8h*)(hc + 32 * s - 16 + (hi ? 8 : 0));
    u.h[1] = *(const v8h*)(hc + 32 * s + (hi ? 8 : 0));
    ak[s] = u.v;
  }

  // V = target_feat @ Wv + bv ; scatter att*V ; stage V (f16) for the MLP
#pragma unroll
  for (int j = 0; j < 8; j++) {
    v8f va = {};
    const _Float16* wvrow = WvT + (16 * j + m) * KVK;
#pragma unroll
    for (int s = 0; s < 5; s++) va = wmma_f16(ak[s], loadB(wvrow, s, hi), va);
    int n = 16 * j + m;
    float bvn = bkv[2 * n + 1];
#pragma unroll
    for (int r = 0; r < 8; r++) {
      float vw = va[r] + bvn;
      W.buf[(r + 8 * hi) * 128 + n] = (_Float16)vw;
      atomicAdd(&agg_h[(size_t)rowsv[r] * 128 + n], attv[r] * vw);
    }
  }
  __builtin_amdgcn_wave_barrier();

  // V as A-fragments for layer 1
  v16h vaf[4];
  const _Float16* vrow = &W.buf[m * 128];
#pragma unroll
  for (int s = 0; s < 4; s++) vaf[s] = loadA(vrow + 32 * s, hi);
  __builtin_amdgcn_wave_barrier();

  // H1 = silu(V @ Wc1)  (256 wide), staged back to LDS as f16
#pragma unroll
  for (int jj = 0; jj < 16; jj++) {
    v8f ha = {};
    const _Float16* w1row = Wc1T + (16 * jj + m) * 128;
#pragma unroll
    for (int s = 0; s < 4; s++) ha = wmma_f16(vaf[s], loadB(w1row, s, hi), ha);
    int n2 = 16 * jj + m;
#pragma unroll
    for (int r = 0; r < 8; r++) {
      float x = ha[r];
      float sl = x / (1.f + __expf(-x));
      W.buf[(r + 8 * hi) * 256 + n2] = (_Float16)sl;
    }
  }
  __builtin_amdgcn_wave_barrier();

  // coord_v = att * (H1 @ Wc2)   (N padded 4 -> 16, zero cols harmless)
  v8f ca = {};
  const _Float16* h1row = &W.buf[m * 256];
  const _Float16* w2row = Wc2T + m * 256;
#pragma unroll
  for (int s = 0; s < 8; s++) {
    v16h a2 = loadA(h1row + 32 * s, hi);
    ca = wmma_f16(a2, loadB(w2row, s, hi), ca);
  }
  if (m < 4) {
#pragma unroll
    for (int r = 0; r < 8; r++) W.cv[(r + 8 * hi) * 4 + m] = attv[r] * ca[r];
  }
  __builtin_amdgcn_wave_barrier();

  // trans = coord_diff * coord_v ; scatter into agg_c
#pragma unroll
  for (int a2i = 0; a2i < 2; a2i++) {
    int c = 2 * hi + a2i;
    float cvv = W.cv[m * 4 + c];
#pragma unroll
    for (int d = 0; d < 3; d++)
      atomicAdd(&agg_c[((size_t)ridx * 4 + c) * 3 + d], df[c * 3 + d] * cvv);
  }
}

// ---------------- finalize ----------------
__global__ void k_final_h(const float* __restrict__ h, const float* __restrict__ agg_h,
                          float* __restrict__ out_h) {
  int i = blockIdx.x * 256 + threadIdx.x;
  if (i < N_NODES * IN_NF) out_h[i] = h[i] + agg_h[i];
}
__global__ void k_final_c(const float* __restrict__ coord, const float* __restrict__ agg_c,
                          float* __restrict__ out_c) {
  int i = blockIdx.x * 256 + threadIdx.x;
  if (i < N_NODES * 12) {
    float a = agg_c[i];
    a = fminf(COORD_CLAMP, fmaxf(-COORD_CLAMP, a));
    out_c[i] = coord[i] + a;
  }
}

// ---------------- launch ----------------
extern "C" void kernel_launch(void* const* d_in, const int* in_sizes, int n_in,
                              void* d_out, int out_size, void* d_ws, size_t ws_size,
                              hipStream_t stream) {
  const float* h     = (const float*)d_in[0];
  const float* coord = (const float*)d_in[1];
  const float* Wq    = (const float*)d_in[2];
  const float* bq    = (const float*)d_in[3];
  const float* Wkv   = (const float*)d_in[4];
  const float* bkv   = (const float*)d_in[5];
  const float* Wc1   = (const float*)d_in[6];
  const float* Wc2   = (const float*)d_in[7];
  const int*   ei    = (const int*)d_in[8];

  char* w = (char*)d_ws;
  size_t off = 0;
  auto carve = [&](size_t bytes) -> char* {
    char* p = w + off;
    off += (bytes + 255) & ~(size_t)255;
    return p;
  };
  float*     alphabuf = (float*)carve((size_t)N_EDGES * 4);
  float*     nodemax  = (float*)carve((size_t)N_NODES * 4);
  float*     nodesum  = (float*)carve((size_t)N_NODES * 4);
  float*     agg_h    = (float*)carve((size_t)N_NODES * 128 * 4);
  float*     agg_c    = (float*)carve((size_t)N_NODES * 12 * 4);
  _Float16*  hbf      = (_Float16*)carve((size_t)N_NODES * KPAD * 2);
  _Float16*  WqT      = (_Float16*)carve((size_t)128 * 128 * 2);
  _Float16*  WkT      = (_Float16*)carve((size_t)128 * KVK * 2);
  _Float16*  WvT      = (_Float16*)carve((size_t)128 * KVK * 2);
  _Float16*  Wc1T     = (_Float16*)carve((size_t)256 * 128 * 2);
  _Float16*  Wc2T     = (_Float16*)carve((size_t)16 * 256 * 2);

  float* out_h   = (float*)d_out;
  float* out_c   = out_h + (size_t)N_NODES * IN_NF;
  float* out_att = out_c + (size_t)N_NODES * 12;

  // prep (stream-ordered, so later kernels see results)
  k_init_nodes<<<(N_NODES + 255) / 256, 256, 0, stream>>>(nodemax, nodesum);
  k_zero<<<(N_NODES * 128 + 255) / 256, 256, 0, stream>>>(agg_h, N_NODES * 128);
  k_zero<<<(N_NODES * 12 + 255) / 256, 256, 0, stream>>>(agg_c, N_NODES * 12);
  k_hpad<<<(N_NODES * KPAD + 255) / 256, 256, 0, stream>>>(h, hbf);
  k_wq<<<(128 * 128 + 255) / 256, 256, 0, stream>>>(Wq, WqT);
  k_wkv<<<(128 * KVK + 255) / 256, 256, 0, stream>>>(Wkv, WkT, WvT);
  k_wc1<<<(256 * 128 + 255) / 256, 256, 0, stream>>>(Wc1, Wc1T);
  k_wc2<<<(16 * 256 + 255) / 256, 256, 0, stream>>>(Wc2, Wc2T);

  // attention pipeline
  k_alpha<<<N_EDGES / (16 * 8), 256, 0, stream>>>(ei, coord, hbf, WqT, WkT, bq, bkv,
                                                  alphabuf, nodemax);
  k_softmax_ex<<<(N_EDGES + 255) / 256, 256, 0, stream>>>(ei, alphabuf, nodemax, nodesum);
  k_aggregate<<<N_EDGES / (16 * 4), 128, 0, stream>>>(ei, coord, hbf, WvT, Wc1T, Wc2T, bkv,
                                                      alphabuf, nodesum, agg_h, agg_c, out_att);

  // finalize outputs
  k_final_h<<<(N_NODES * 128 + 255) / 256, 256, 0, stream>>>(h, agg_h, out_h);
  k_final_c<<<(N_NODES * 12 + 255) / 256, 256, 0, stream>>>(coord, agg_c, out_c);
}
